// AttnG2P_21509196218945
// MI455X (gfx1250) — matrix operand order repaired and checked by reading, not compiled, not run on hardware
//
#include <hip/hip_runtime.h>
#include <hip/hip_bf16.h>
#include <math.h>

typedef _Float16 h16;
typedef __attribute__((ext_vector_type(16))) _Float16 v16h;
typedef __attribute__((ext_vector_type(8)))  _Float16 v8h;
typedef __attribute__((ext_vector_type(8)))  float    v8f;
typedef __attribute__((ext_vector_type(4)))  int      v4i_t;

#define AS1 __attribute__((address_space(1)))
#define AS3 __attribute__((address_space(3)))

#define HDIM 512
#define BDIM 256
#define SDIM 48
#define TDIM 48
#define VDIM 200
#define VPAD 208
#define LP   40   // LDS row pitch in halfs (32 data + 8 pad -> 80B, bank-spread)

#if __has_builtin(__builtin_amdgcn_global_load_async_to_lds_b128) && __has_builtin(__builtin_amdgcn_s_wait_asynccnt)
#define LDS_ASYNC 1
#else
#define LDS_ASYNC 0
#endif

// ---------------------------------------------------------------------------
// Tile loaders (contiguous k order used by this file, not the ISA table order; wave32)
// A 16x32 f16: lane<16 -> row=lane, K k0..k0+15 contiguous
//              lane>=16 -> row=lane-16, K k0+16..k0+31 contiguous
// B 32x16 f16: lane<16 -> col=lane, K k0..k0+15 contiguous
//              lane>=16 -> col=lane-16, K k0+16..k0+31 contiguous
// ---------------------------------------------------------------------------
__device__ __forceinline__ v16h load_a_tile(const h16* __restrict__ rowptr, int k0, int hi) {
    v16h a;
    const h16* p = rowptr + k0 + hi * 16;
    *(v8h*)&a         = *(const v8h*)p;
    *(((v8h*)&a) + 1) = *(const v8h*)(p + 8);
    return a;
}
__device__ __forceinline__ v16h load_b_tile(const h16* __restrict__ rowptr, int k0, int hi) {
    return *(const v16h*)(rowptr + k0 + hi * 16);
}
__device__ __forceinline__ float sigmf(float x) { return 1.0f / (1.0f + __expf(-x)); }

// stage one 16-byte chunk of a weight tile into LDS (async on CDNA5)
__device__ __forceinline__ void stage_one(const h16* src, h16* dst) {
#if LDS_ASYNC
    __builtin_amdgcn_global_load_async_to_lds_b128(
        (AS1 v4i_t*)src, (AS3 v4i_t*)dst, 0, 0);
#else
    *(v8h*)dst = *(const v8h*)src;
#endif
}

__device__ __forceinline__ const h16* stage_src(int it, int nx, const h16* Wih, int KX,
                                                const h16* Whh, int j0,
                                                int sg, int sr, int schunk) {
    if (it < nx) return Wih + (long)(sg * HDIM + j0 + sr) * KX  + it * 32 + schunk;
    return        Whh + (long)(sg * HDIM + j0 + sr) * HDIM + (it - nx) * 32 + schunk;
}

// ---------------------------------------------------------------------------
// Fused LSTM step with async LDS weight staging:
//   gates = X@Wih^T + Hin@Whh^T + bias; c,h elementwise update; h out f16.
// One block (8 waves) per 16-wide h-column tile (jt). Per K-slice, the 4-gate
// weight tiles (64 rows x 32 halfs) are async-copied to LDS once and consumed
// by all 8 waves; each wave owns TWO 16-row batch tiles, so the 4 LDS B-tiles
// feed 8 WMMAs (2x arithmetic intensity per DS read). Double-buffered.
// Grid: 32 blocks x 256 thr; jt = blockIdx.x, wave covers batches wv*32..+31.
// If tok != nullptr, A rows for the X part are gathered: X[tok[b*ts+to]].
// ---------------------------------------------------------------------------
__global__ void lstm_step_kernel(const h16* __restrict__ X, int ldx, int KX,
                                 const int* __restrict__ tok, int tok_stride, int tok_off,
                                 const h16* __restrict__ Wih, const h16* __restrict__ Whh,
                                 const float* __restrict__ bias,
                                 const h16* __restrict__ h_in, float* __restrict__ c,
                                 h16* __restrict__ h_out, h16* __restrict__ seq_out)
{
    __shared__ h16 wbuf[2][64 * LP];

    const int t    = threadIdx.x;
    const int lane = t & 31;
    const int wv   = t >> 5;
    const int jt   = blockIdx.x;          // 0..31
    const int b0   = wv * 32;             // two batch tiles: rows b0..b0+31
    const int j0   = jt * 16;
    const int hi = (lane >> 4) & 1, lm = lane & 15;
    const int jcol = j0 + lm;
    const int nx   = KX >> 5;
    const int ntot = nx + (HDIM >> 5);

    // per-thread staging slot: 64 rows (4 gates x 16 weight rows) x 4 chunks of 8 halfs
    const int srow = t >> 2;
    const int schunk = (t & 3) * 8;
    const int sg = srow >> 4, sr = srow & 15;

    v8f acc0[4], acc1[4];
#pragma unroll
    for (int g = 0; g < 4; ++g) {
        float bv = bias[g * HDIM + jcol];
#pragma unroll
        for (int r = 0; r < 8; ++r) { acc0[g][r] = bv; acc1[g][r] = bv; }
    }

    const int arow0 = b0 + lm;
    const int arow1 = b0 + 16 + lm;
    const h16* aptr0 = tok ? (X + (long)tok[arow0 * tok_stride + tok_off] * ldx)
                           : (X + (long)arow0 * ldx);
    const h16* aptr1 = tok ? (X + (long)tok[arow1 * tok_stride + tok_off] * ldx)
                           : (X + (long)arow1 * ldx);
    const h16* hptr0 = h_in + (long)arow0 * HDIM;
    const h16* hptr1 = h_in + (long)arow1 * HDIM;

    // prologue: stage K-slice 0
    stage_one(stage_src(0, nx, Wih, KX, Whh, j0, sg, sr, schunk),
              &wbuf[0][srow * LP + schunk]);

    for (int it = 0; it < ntot; ++it) {
        if (it + 1 < ntot) {
            stage_one(stage_src(it + 1, nx, Wih, KX, Whh, j0, sg, sr, schunk),
                      &wbuf[(it + 1) & 1][srow * LP + schunk]);
#if LDS_ASYNC
            __builtin_amdgcn_s_wait_asynccnt(1);   // slice `it` landed; it+1 in flight
#endif
        } else {
#if LDS_ASYNC
            __builtin_amdgcn_s_wait_asynccnt(0);
#endif
        }
        __syncthreads();                           // slice `it` visible to all waves

        const h16 *ap0, *ap1; int k0;
        if (it < nx) { ap0 = aptr0; ap1 = aptr1; k0 = it * 32; }
        else         { ap0 = hptr0; ap1 = hptr1; k0 = (it - nx) * 32; }
        v16h a0 = load_a_tile(ap0, k0, hi);
        v16h a1 = load_a_tile(ap1, k0, hi);

        const h16* lbase = &wbuf[it & 1][0];
        v16h bm0 = *(const v16h*)(lbase + (0 * 16 + lm) * LP + hi * 16);
        v16h bm1 = *(const v16h*)(lbase + (1 * 16 + lm) * LP + hi * 16);
        v16h bm2 = *(const v16h*)(lbase + (2 * 16 + lm) * LP + hi * 16);
        v16h bm3 = *(const v16h*)(lbase + (3 * 16 + lm) * LP + hi * 16);

        acc0[0] = __builtin_amdgcn_wmma_f32_16x16x32_f16(false, a0, false, bm0, (short)0, acc0[0], false, false);
        acc1[0] = __builtin_amdgcn_wmma_f32_16x16x32_f16(false, a1, false, bm0, (short)0, acc1[0], false, false);
        acc0[1] = __builtin_amdgcn_wmma_f32_16x16x32_f16(false, a0, false, bm1, (short)0, acc0[1], false, false);
        acc1[1] = __builtin_amdgcn_wmma_f32_16x16x32_f16(false, a1, false, bm1, (short)0, acc1[1], false, false);
        acc0[2] = __builtin_amdgcn_wmma_f32_16x16x32_f16(false, a0, false, bm2, (short)0, acc0[2], false, false);
        acc1[2] = __builtin_amdgcn_wmma_f32_16x16x32_f16(false, a1, false, bm2, (short)0, acc1[2], false, false);
        acc0[3] = __builtin_amdgcn_wmma_f32_16x16x32_f16(false, a0, false, bm3, (short)0, acc0[3], false, false);
        acc1[3] = __builtin_amdgcn_wmma_f32_16x16x32_f16(false, a1, false, bm3, (short)0, acc1[3], false, false);

        __syncthreads();                           // all waves done before buffer reuse
    }

    // cell update: tile0 rows b0+hi*8+r, tile1 rows b0+16+hi*8+r, column jcol
#pragma unroll
    for (int r = 0; r < 8; ++r) {
        {
            int b = b0 + hi * 8 + r;
            long idx = (long)b * HDIM + jcol;
            float iv = sigmf(acc0[0][r]);
            float fv = sigmf(acc0[1][r]);
            float gv = tanhf(acc0[2][r]);
            float ov = sigmf(acc0[3][r]);
            float cn = fv * c[idx] + iv * gv;
            c[idx] = cn;
            h16 hh = (h16)(ov * tanhf(cn));
            h_out[idx] = hh;
            if (seq_out) seq_out[idx] = hh;
        }
        {
            int b = b0 + 16 + hi * 8 + r;
            long idx = (long)b * HDIM + jcol;
            float iv = sigmf(acc1[0][r]);
            float fv = sigmf(acc1[1][r]);
            float gv = tanhf(acc1[2][r]);
            float ov = sigmf(acc1[3][r]);
            float cn = fv * c[idx] + iv * gv;
            c[idx] = cn;
            h16 hh = (h16)(ov * tanhf(cn));
            h_out[idx] = hh;
            if (seq_out) seq_out[idx] = hh;
        }
    }
}

// ---------------------------------------------------------------------------
// Generic WMMA GEMM: C[m,n] = sum_k A[m,k]*W[n,k] + bias[n]; M fixed = 256.
// One wave per 16x16 output tile.
// ---------------------------------------------------------------------------
__global__ void gemm_bias_kernel(const h16* __restrict__ A, int lda,
                                 const h16* __restrict__ W, int ldw, int K,
                                 const float* __restrict__ bias,
                                 float* __restrict__ C, int ldc, int ntiles)
{
    int wid = (blockIdx.x * blockDim.x + threadIdx.x) >> 5;
    if (wid >= (BDIM / 16) * ntiles) return;
    int lane = threadIdx.x & 31;
    int mt = wid / ntiles, nt = wid % ntiles;
    int hi = (lane >> 4) & 1, lm = lane & 15;
    int ncol = nt * 16 + lm;

    v8f acc;
    float bv = bias ? bias[ncol] : 0.0f;
#pragma unroll
    for (int r = 0; r < 8; ++r) acc[r] = bv;

    const h16* aptr = A + (long)(mt * 16 + lm) * lda;
    const h16* wptr = W + (long)ncol * ldw;
    for (int k0 = 0; k0 < K; k0 += 32) {
        v16h a  = load_a_tile(aptr, k0, hi);
        v16h bm = load_b_tile(wptr, k0, hi);
        acc = __builtin_amdgcn_wmma_f32_16x16x32_f16(false, a, false, bm,
                                                     (short)0, acc, false, false);
    }
#pragma unroll
    for (int r = 0; r < 8; ++r) {
        int m = mt * 16 + hi * 8 + r;
        C[(long)m * ldc + ncol] = acc[r];
    }
}

// ---------------------------------------------------------------------------
// Decoder attention: one block (256 thr) per batch row.
// scores = enc_out . q * scale, masked softmax, attn -> xt_dec[:,0:H] (f16),
// emb(y_in) -> xt_dec[:,H:2H].
// ---------------------------------------------------------------------------
__global__ void attn_kernel(const h16* __restrict__ enc_out,   // [S,B,H]
                            const float* __restrict__ q,       // [B,H]
                            const int* __restrict__ x,         // [B,S]
                            const h16* __restrict__ dec_embed, // [V,H] f16
                            const int* __restrict__ y,         // [B,T]
                            int step, h16* __restrict__ xt_dec) // [B,2H]
{
    __shared__ float qs[HDIM];
    __shared__ float sc[SDIM];
    int b = blockIdx.x, t = threadIdx.x;
    for (int h = t; h < HDIM; h += 256) qs[h] = q[(long)b * HDIM + h];
    __syncthreads();

    int w = t >> 5, lane = t & 31;
    const float scale = 0.044194173824159216f;   // 1/sqrt(512)
    for (int si = 0; si < 6; ++si) {
        int s = w * 6 + si;
        float partial = 0.0f;
        const h16* e = enc_out + (long)s * BDIM * HDIM + (long)b * HDIM;
        for (int kk = 0; kk < HDIM / 32; ++kk) {
            int h = lane + kk * 32;
            partial += (float)e[h] * qs[h];
        }
#pragma unroll
        for (int off = 16; off > 0; off >>= 1) partial += __shfl_xor(partial, off, 32);
        if (lane == 0) sc[s] = (x[b * SDIM + s] != 0) ? partial * scale : -INFINITY;
    }
    __syncthreads();
    if (t == 0) {
        float m = -INFINITY;
        for (int s = 0; s < SDIM; ++s) m = fmaxf(m, sc[s]);
        float sum = 0.0f;
        for (int s = 0; s < SDIM; ++s) sum += __expf(sc[s] - m);
        float inv = 1.0f / sum;
        for (int s = 0; s < SDIM; ++s) sc[s] = __expf(sc[s] - m) * inv;
    }
    __syncthreads();
    int tokin = (step == 0) ? 1 : y[b * TDIM + (step - 1)];
    for (int h = t; h < HDIM; h += 256) {
        float a = 0.0f;
        const h16* e = enc_out + (long)b * HDIM + h;
        for (int s = 0; s < SDIM; ++s) a += sc[s] * (float)e[(long)s * BDIM * HDIM];
        xt_dec[(long)b * 2 * HDIM + h]        = (h16)a;
        xt_dec[(long)b * 2 * HDIM + HDIM + h] = dec_embed[(long)tokin * HDIM + h];
    }
}

// q input build: xt_q = [emb(y_in), h_dec] f16
__global__ void build_qin_kernel(const h16* __restrict__ dec_embed, const int* __restrict__ y,
                                 int step, const h16* __restrict__ h_dec, h16* __restrict__ xt_q)
{
    int b = blockIdx.x, t = threadIdx.x;
    int tokin = (step == 0) ? 1 : y[b * TDIM + (step - 1)];
    for (int h = t; h < HDIM; h += 256) {
        xt_q[(long)b * 2 * HDIM + h]        = dec_embed[(long)tokin * HDIM + h];
        xt_q[(long)b * 2 * HDIM + HDIM + h] = h_dec[(long)b * HDIM + h];
    }
}

// per-step masked-mean NLL from logits (log-softmax over first VDIM cols)
__global__ void loss_kernel(const float* __restrict__ logits, const int* __restrict__ y,
                            int step, float* __restrict__ loss_steps)
{
    __shared__ float s_nll[256];
    __shared__ float s_val[256];
    int b = threadIdx.x;
    const float* L = logits + (long)b * VPAD;
    float m = L[0];
    for (int v = 1; v < VDIM; ++v) m = fmaxf(m, L[v]);
    float sum = 0.0f;
    for (int v = 0; v < VDIM; ++v) sum += __expf(L[v] - m);
    float lse = m + __logf(sum);
    int yo = (step < TDIM) ? y[b * TDIM + step] : 1;
    float val = (yo != 0) ? 1.0f : 0.0f;
    s_nll[b] = (lse - L[yo]) * val;
    s_val[b] = val;
    __syncthreads();
    for (int off = 128; off > 0; off >>= 1) {
        if (b < off) { s_nll[b] += s_nll[b + off]; s_val[b] += s_val[b + off]; }
        __syncthreads();
    }
    if (b == 0) loss_steps[step] = s_nll[0] / fmaxf(s_val[0], 1.0f);
}

__global__ void final_sum_kernel(const float* __restrict__ ls, int n, float* __restrict__ out)
{
    if (threadIdx.x == 0) {
        float s = 0.0f;
        for (int i = 0; i < n; ++i) s += ls[i];
        out[0] = s;
    }
}

// ---------------------------------------------------------------------------
// small utility kernels
// ---------------------------------------------------------------------------
__global__ void cvt_f32_f16_kernel(const float* __restrict__ in, h16* __restrict__ out, long n) {
    long i = (long)blockIdx.x * blockDim.x + threadIdx.x;
    if (i < n) out[i] = (h16)in[i];
}
__global__ void zero_f32_kernel(float* p, long n) {
    long i = (long)blockIdx.x * blockDim.x + threadIdx.x;
    if (i < n) p[i] = 0.0f;
}
__global__ void zero_f16_kernel(h16* p, long n) {
    long i = (long)blockIdx.x * blockDim.x + threadIdx.x;
    if (i < n) p[i] = (h16)0.0f;
}
__global__ void add2_f32_kernel(const float* a, const float* b, float* o, long n) {
    long i = (long)blockIdx.x * blockDim.x + threadIdx.x;
    if (i < n) o[i] = a[i] + b[i];
}
__global__ void copy_f32_kernel(const float* a, float* o, long n) {
    long i = (long)blockIdx.x * blockDim.x + threadIdx.x;
    if (i < n) o[i] = a[i];
}

// ---------------------------------------------------------------------------
static inline char* ws_take(char*& p, size_t bytes) {
    char* r = p;
    p += (bytes + 255) & ~(size_t)255;
    return r;
}
static inline int gdiv(long n, int b) { return (int)((n + b - 1) / b); }

extern "C" void kernel_launch(void* const* d_in, const int* in_sizes, int n_in,
                              void* d_out, int out_size, void* d_ws, size_t ws_size,
                              hipStream_t stream) {
    // inputs (setup_inputs order)
    const int*   x        = (const int*)d_in[0];    // [B,S]
    const int*   y        = (const int*)d_in[1];    // [B,T]
    const float* enc_embed= (const float*)d_in[2];
    const float* enc_Wih0 = (const float*)d_in[3];
    const float* enc_Whh0 = (const float*)d_in[4];
    const float* enc_bih0 = (const float*)d_in[5];
    const float* enc_bhh0 = (const float*)d_in[6];
    const float* enc_Wih1 = (const float*)d_in[7];
    const float* enc_Whh1 = (const float*)d_in[8];
    const float* enc_bih1 = (const float*)d_in[9];
    const float* enc_bhh1 = (const float*)d_in[10];
    const float* dec_embed= (const float*)d_in[11];
    const float* dec_Wih  = (const float*)d_in[12]; // [4H, 2H]
    const float* dec_Whh  = (const float*)d_in[13]; // [4H, H]
    const float* dec_bih  = (const float*)d_in[14];
    const float* dec_bhh  = (const float*)d_in[15];
    const float* linQ_W   = (const float*)d_in[16]; // [H, 2H]
    const float* linQ_b   = (const float*)d_in[17]; // [H]
    const float* out_W    = (const float*)d_in[18]; // [V, H]
    const float* out_b    = (const float*)d_in[19]; // [V]
    float* out = (float*)d_out;

    // workspace layout
    char* p = (char*)d_ws;
    const long EMB   = (long)VDIM * HDIM;
    const long W4H_H = (long)4 * HDIM * HDIM;
    const long W4H_2H= (long)4 * HDIM * 2 * HDIM;
    const long BH    = (long)BDIM * HDIM;
    const long SBH   = (long)SDIM * BDIM * HDIM;

    h16* encEmb_h = (h16*)ws_take(p, EMB * 2);
    h16* decEmb_h = (h16*)ws_take(p, EMB * 2);
    h16* Wih0_h   = (h16*)ws_take(p, W4H_H * 2);
    h16* Whh0_h   = (h16*)ws_take(p, W4H_H * 2);
    h16* Wih1_h   = (h16*)ws_take(p, W4H_H * 2);
    h16* Whh1_h   = (h16*)ws_take(p, W4H_H * 2);
    h16* dWih_h   = (h16*)ws_take(p, W4H_2H * 2);
    h16* dWhh_h   = (h16*)ws_take(p, W4H_H * 2);
    h16* linQ_h   = (h16*)ws_take(p, (long)HDIM * 2 * HDIM * 2);
    h16* outW_h   = (h16*)ws_take(p, (long)VPAD * HDIM * 2);
    float* b0c    = (float*)ws_take(p, 4 * HDIM * 4);
    float* b1c    = (float*)ws_take(p, 4 * HDIM * 4);
    float* bdc    = (float*)ws_take(p, 4 * HDIM * 4);
    float* outb_p = (float*)ws_take(p, VPAD * 4);
    h16* seq1     = (h16*)ws_take(p, SBH * 2);
    h16* enc_out  = (h16*)ws_take(p, SBH * 2);
    h16* h0a      = (h16*)ws_take(p, BH * 2);
    h16* h0b      = (h16*)ws_take(p, BH * 2);
    h16* h1a      = (h16*)ws_take(p, BH * 2);
    h16* h1b      = (h16*)ws_take(p, BH * 2);
    h16* hda      = (h16*)ws_take(p, BH * 2);
    h16* hdb      = (h16*)ws_take(p, BH * 2);
    float* c0     = (float*)ws_take(p, BH * 4);
    float* c1     = (float*)ws_take(p, BH * 4);
    float* cd     = (float*)ws_take(p, BH * 4);
    h16* xt_q     = (h16*)ws_take(p, (long)BDIM * 2 * HDIM * 2);
    h16* xt_dec   = (h16*)ws_take(p, (long)BDIM * 2 * HDIM * 2);
    float* qbuf   = (float*)ws_take(p, BH * 4);
    float* logits = (float*)ws_take(p, (long)BDIM * VPAD * 4);
    float* lsteps = (float*)ws_take(p, 64 * 4);
    (void)ws_size; (void)n_in; (void)in_sizes; (void)out_size;

    // ---- weight conversion / prep ----
    cvt_f32_f16_kernel<<<gdiv(EMB,256),256,0,stream>>>(enc_embed, encEmb_h, EMB);
    cvt_f32_f16_kernel<<<gdiv(EMB,256),256,0,stream>>>(dec_embed, decEmb_h, EMB);
    cvt_f32_f16_kernel<<<gdiv(W4H_H,256),256,0,stream>>>(enc_Wih0, Wih0_h, W4H_H);
    cvt_f32_f16_kernel<<<gdiv(W4H_H,256),256,0,stream>>>(enc_Whh0, Whh0_h, W4H_H);
    cvt_f32_f16_kernel<<<gdiv(W4H_H,256),256,0,stream>>>(enc_Wih1, Wih1_h, W4H_H);
    cvt_f32_f16_kernel<<<gdiv(W4H_H,256),256,0,stream>>>(enc_Whh1, Whh1_h, W4H_H);
    cvt_f32_f16_kernel<<<gdiv(W4H_2H,256),256,0,stream>>>(dec_Wih, dWih_h, W4H_2H);
    cvt_f32_f16_kernel<<<gdiv(W4H_H,256),256,0,stream>>>(dec_Whh, dWhh_h, W4H_H);
    cvt_f32_f16_kernel<<<gdiv((long)HDIM*2*HDIM,256),256,0,stream>>>(linQ_W, linQ_h, (long)HDIM*2*HDIM);
    zero_f16_kernel<<<gdiv((long)VPAD*HDIM,256),256,0,stream>>>(outW_h, (long)VPAD*HDIM);
    cvt_f32_f16_kernel<<<gdiv((long)VDIM*HDIM,256),256,0,stream>>>(out_W, outW_h, (long)VDIM*HDIM);
    zero_f32_kernel<<<1,256,0,stream>>>(outb_p, VPAD);
    copy_f32_kernel<<<1,256,0,stream>>>(out_b, outb_p, VDIM);
    add2_f32_kernel<<<gdiv(4*HDIM,256),256,0,stream>>>(enc_bih0, enc_bhh0, b0c, 4*HDIM);
    add2_f32_kernel<<<gdiv(4*HDIM,256),256,0,stream>>>(enc_bih1, enc_bhh1, b1c, 4*HDIM);
    add2_f32_kernel<<<gdiv(4*HDIM,256),256,0,stream>>>(dec_bih, dec_bhh, bdc, 4*HDIM);
    // zero states
    zero_f16_kernel<<<gdiv(BH,256),256,0,stream>>>(h0a, BH);
    zero_f16_kernel<<<gdiv(BH,256),256,0,stream>>>(h0b, BH);
    zero_f16_kernel<<<gdiv(BH,256),256,0,stream>>>(h1a, BH);
    zero_f16_kernel<<<gdiv(BH,256),256,0,stream>>>(h1b, BH);
    zero_f16_kernel<<<gdiv(BH,256),256,0,stream>>>(hda, BH);
    zero_f16_kernel<<<gdiv(BH,256),256,0,stream>>>(hdb, BH);
    zero_f32_kernel<<<gdiv(BH,256),256,0,stream>>>(c0, BH);
    zero_f32_kernel<<<gdiv(BH,256),256,0,stream>>>(c1, BH);
    zero_f32_kernel<<<gdiv(BH,256),256,0,stream>>>(cd, BH);

    const int LSTM_BLOCKS = 32;   // one block per h-column tile; 8 waves x 2 batch tiles

    // ---- encoder layer 0 (embedding gathered in A-load) ----
    {
        h16 *hin = h0a, *hout = h0b;
        for (int s = 0; s < SDIM; ++s) {
            lstm_step_kernel<<<LSTM_BLOCKS,256,0,stream>>>(
                encEmb_h, HDIM, HDIM, x, SDIM, s,
                Wih0_h, Whh0_h, b0c, hin, c0, hout, seq1 + (long)s * BDIM * HDIM);
            h16* t = hin; hin = hout; hout = t;
        }
    }
    // ---- encoder layer 1 ----
    {
        h16 *hin = h1a, *hout = h1b;
        for (int s = 0; s < SDIM; ++s) {
            lstm_step_kernel<<<LSTM_BLOCKS,256,0,stream>>>(
                seq1 + (long)s * BDIM * HDIM, HDIM, HDIM, nullptr, 0, 0,
                Wih1_h, Whh1_h, b1c, hin, c1, hout, enc_out + (long)s * BDIM * HDIM);
            h16* t = hin; hin = hout; hout = t;
        }
    }
    // ---- decoder: T+1 steps ----
    {
        h16 *hin = hda, *hout = hdb;
        for (int t = 0; t < TDIM + 1; ++t) {
            build_qin_kernel<<<BDIM,256,0,stream>>>(decEmb_h, y, t, hin, xt_q);
            // q = xt_q @ linQ_W^T + linQ_b : [256 x 512], K=1024 -> 512 waves
            gemm_bias_kernel<<<64,256,0,stream>>>(xt_q, 2*HDIM, linQ_h, 2*HDIM, 2*HDIM,
                                                  linQ_b, qbuf, HDIM, HDIM/16);
            attn_kernel<<<BDIM,256,0,stream>>>(enc_out, qbuf, x, decEmb_h, y, t, xt_dec);
            lstm_step_kernel<<<LSTM_BLOCKS,256,0,stream>>>(
                xt_dec, 2*HDIM, 2*HDIM, nullptr, 0, 0,
                dWih_h, dWhh_h, bdc, hin, cd, hout, nullptr);
            // logits = h @ out_W^T + out_b : [256 x 208], K=512 -> 208 waves -> 26 blocks
            gemm_bias_kernel<<<26,256,0,stream>>>(hout, HDIM, outW_h, HDIM, HDIM,
                                                  outb_p, logits, VPAD, VPAD/16);
            loss_kernel<<<1,256,0,stream>>>(logits, y, t, lsteps);
            h16* tmp = hin; hin = hout; hout = tmp;
        }
    }
    final_sum_kernel<<<1,64,0,stream>>>(lsteps, TDIM + 1, out);
}
